// GroupedQueryAttention_45114336477259
// MI455X (gfx1250) — compile-verified, matrix operand order checked by reading, not run on hardware
//
#include <hip/hip_runtime.h>
#include <math.h>

// ---------------------------------------------------------------------------
// GQA attention block for MI455X (gfx1250), wave32 + WMMA bf16 16x16x32.
// Pipeline: [qkv gemm (TDM->LDS->WMMA)] -> [rope] -> [flash attention (WMMA)]
//           -> [out gemm (TDM->LDS->WMMA)]
// GEMM tiles are staged into LDS by the Tensor Data Mover (tensor_load_to_lds,
// TENSORcnt), double-buffered so the DMA of tile k+1 overlaps WMMA on tile k.
// ---------------------------------------------------------------------------

typedef __attribute__((ext_vector_type(16))) __bf16 v16bf;
typedef __attribute__((ext_vector_type(8)))  float  v8f;
typedef __attribute__((ext_vector_type(4)))  unsigned int u32x4;
typedef __attribute__((ext_vector_type(8)))  int  i32x8;
typedef __attribute__((ext_vector_type(4)))  int  i32x4;

#define N_HEAD   16
#define N_KVHEAD 4
#define HS       128
#define QKV_LD   3072      // (16+4+4)*128
#define MODEL    2048

__device__ __forceinline__ v8f wmma_bf16(v16bf a, v16bf b, v8f c) {
  // (neg_a, A, neg_b, B, c_mod, C, reuse_a, reuse_b)
  return __builtin_amdgcn_wmma_f32_16x16x32_bf16(false, a, false, b, (short)0, c,
                                                 false, false);
}

// Load a 16x32 fp32 tile (row stride ld floats) as a bf16 A/B fragment.
// CDNA5 16-bit 16x32 fragment layout:
//   lane L<16 : row L, elems 0..7 -> K=0..7,  elems 8..15 -> K=16..23
//   lane L>=16: row L, elems 0..7 -> K=8..15, elems 8..15 -> K=24..31
__device__ __forceinline__ v16bf load_frag(const float* base, int ld, int lane) {
  const int r  = lane & 15;
  const int hh = lane >> 4;
  const float* rowp = base + (long)r * ld;
  const int k0 = hh * 8;
  const int k1 = 16 + hh * 8;
  v16bf f;
#pragma unroll
  for (int i = 0; i < 8; ++i) f[i] = (__bf16)rowp[k0 + i];
#pragma unroll
  for (int i = 0; i < 8; ++i) f[8 + i] = (__bf16)rowp[k1 + i];
  return f;
}

// ---------------------------------------------------------------------------
// Tensor Data Mover: 2D tile (rows x 32 f32) from global into LDS.
// D# packing per CDNA5 ISA 8.3/8.4 (group0: count/lds/global/type,
// group1: data_size, tensor dims, tile dims, dim0 stride). Groups 2/3 zero
// (2D tensor). Issued by one wave; EXEC is ignored by TDM.
// This toolchain exposes the 6-arg builtin form (g0,g1,g2,g3,g4,cpol).
// ---------------------------------------------------------------------------
__device__ __forceinline__ void tdm_load_2d(const float* gsrc, unsigned lds_off,
                                            int rows, int ld /*floats*/) {
  const unsigned long long ga = (unsigned long long)(const void*)gsrc;
  u32x4 g0;
  g0[0] = 1u;                                             // count=1, user mode
  g0[1] = lds_off;                                        // lds_addr
  g0[2] = (unsigned)(ga & 0xffffffffu);                   // global_addr[31:0]
  g0[3] = (unsigned)((ga >> 32) & 0x01ffffffu) | (2u << 30); // [56:32] | type=2

  const unsigned td0 = 32u;               // tensor_dim0 (== tile width)
  const unsigned td1 = (unsigned)rows;    // tensor_dim1 (== tile rows)
  i32x8 g1;
  g1[0] = (int)(2u << 16);                                // data_size = 4 bytes
  g1[1] = (int)((td0 & 0xffffu) << 16);                   // tensor_dim0[15:0]
  g1[2] = (int)((td0 >> 16) | ((td1 & 0xffffu) << 16));   // td0[31:16]|td1[15:0]
  g1[3] = (int)((td1 >> 16) | (32u << 16));               // td1[31:16]|tile_dim0
  g1[4] = (int)(td1 & 0xffffu);                           // tile_dim1 (tile_dim2=0)
  g1[5] = ld;                                             // tensor_dim0_stride[31:0]
  g1[6] = 0;                                              // stride[47:32] | d1s[15:0]
  g1[7] = 0;
  const i32x4 z4 = {0, 0, 0, 0};
  const i32x8 z8 = {0, 0, 0, 0, 0, 0, 0, 0};
  __builtin_amdgcn_tensor_load_to_lds(g0, g1, z4, z4, z8, 0);
}

// ---------------------------------------------------------------------------
// GEMM: C[M,N] = A[M,K] @ B[N,K]^T   (both operands K-contiguous in memory)
// Block: 128 threads (4 waves). Block tile 128x64; wave tile 32x64.
// K is consumed in 32-wide steps staged into double-buffered LDS by the TDM.
// Grid: (N/64, M/128).
// ---------------------------------------------------------------------------
__global__ void gemm_bf16_wmma(const float* __restrict__ A,
                               const float* __restrict__ Bw,
                               float* __restrict__ C,
                               int M, int N, int K) {
  __shared__ float Atile[2][128 * 32];   // 32 KB
  __shared__ float Btile[2][64 * 32];    // 16 KB

  const int lane = threadIdx.x & 31;
  const int wave = threadIdx.x >> 5;                 // 0..3
  const int rowBase = blockIdx.y * 128;
  const int colBase = blockIdx.x * 64;
  const int waveRow = wave * 32;                     // wave's rows in block tile

  v8f acc[2][4];
#pragma unroll
  for (int m = 0; m < 2; ++m)
#pragma unroll
    for (int n = 0; n < 4; ++n) acc[m][n] = (v8f){};

  const int nsteps = K / 32;

  // prologue: DMA tile 0 into buffer 0
  if (wave == 0) {
    tdm_load_2d(A + (long)rowBase * K,
                (unsigned)(unsigned long long)(const void*)&Atile[0][0], 128, K);
    tdm_load_2d(Bw + (long)colBase * K,
                (unsigned)(unsigned long long)(const void*)&Btile[0][0], 64, K);
  }

  for (int s = 0; s < nsteps; ++s) {
    if (wave == 0) __builtin_amdgcn_s_wait_tensorcnt(0);
    __syncthreads();   // tile s visible to all; also fences LDS reads of s-1

    const int cur = s & 1;
    if (wave == 0 && (s + 1) < nsteps) {          // DMA tile s+1 into other buf
      const int kn = (s + 1) * 32;
      tdm_load_2d(A + (long)rowBase * K + kn,
                  (unsigned)(unsigned long long)(const void*)&Atile[cur ^ 1][0],
                  128, K);
      tdm_load_2d(Bw + (long)colBase * K + kn,
                  (unsigned)(unsigned long long)(const void*)&Btile[cur ^ 1][0],
                  64, K);
    }

    const float* As = &Atile[cur][waveRow * 32];
    const float* Bs = &Btile[cur][0];
    v16bf a0 = load_frag(As, 32, lane);
    v16bf a1 = load_frag(As + 16 * 32, 32, lane);
#pragma unroll
    for (int n = 0; n < 4; ++n) {
      v16bf b = load_frag(Bs + n * 16 * 32, 32, lane);
      acc[0][n] = wmma_bf16(a0, b, acc[0][n]);
      acc[1][n] = wmma_bf16(a1, b, acc[1][n]);
    }
  }

  // C layout: VGPR j -> row j (lanes 0-15) / row j+8 (lanes 16-31), col = lane%16
  const int col   = lane & 15;
  const int half8 = (lane >> 4) * 8;
#pragma unroll
  for (int m = 0; m < 2; ++m)
#pragma unroll
    for (int n = 0; n < 4; ++n)
#pragma unroll
      for (int j = 0; j < 8; ++j)
        C[(long)(rowBase + waveRow + m * 16 + half8 + j) * N + colBase + n * 16 +
          col] = acc[m][n][j];
}

// ---------------------------------------------------------------------------
// RoPE applied in place to the Q and K sections of the qkv buffer.
// One thread per rotation pair: (b,t) x 20 heads x 64 pairs.
// ---------------------------------------------------------------------------
__global__ void rope_kernel(float* __restrict__ qkv, int B, int T) {
  const long idx   = (long)blockIdx.x * blockDim.x + threadIdx.x;
  const int  pairs = (N_HEAD + N_KVHEAD) * (HS / 2);  // 20 * 64
  const long bt    = idx / pairs;
  if (bt >= (long)B * T) return;
  const int p    = (int)(idx % pairs);
  const int t    = (int)(bt % T);
  const int head = p >> 6;          // 0..19 (q heads 0..15, then k heads)
  const int i    = p & 63;          // pair index within head dim

  const float inv = __powf(10000.0f, -(2.0f * (float)i) / 128.0f);
  const float ang = (float)t * inv;
  const float c = __cosf(ang), s = __sinf(ang);

  float* base = qkv + bt * QKV_LD + head * HS;  // q heads then k heads contiguous
  const float u0 = base[i];
  const float u1 = base[i + 64];
  base[i]      = u0 * c - u1 * s;   // u*cos + rotate_half(u)*sin
  base[i + 64] = u1 * c + u0 * s;
}

// ---------------------------------------------------------------------------
// Flash attention, causal, GQA. One wave per (b, q-head, 16-query tile).
// Key blocks of 32: scores = two 16x16 C tiles; P (16x32) -> one bf16
// A-fragment (via LDS layout shuffle) for the PV WMMA.
// Grid: (T/16, N_HEAD, B), block = 32 threads (one wave).
// ---------------------------------------------------------------------------
__global__ void attn_wmma_kernel(const float* __restrict__ qkv,
                                 float* __restrict__ y,   // [B,T,16,128]
                                 int B, int T) {
  __shared__ __bf16 psh[16 * 32];

  const int lane = threadIdx.x;
  const int qt   = blockIdx.x;
  const int h    = blockIdx.y;
  const int b    = blockIdx.z;
  const int kvh  = h >> 2;                 // GQA: 4 query heads per kv head

  const float* qbase = qkv + ((long)b * T + (long)qt * 16) * QKV_LD + h * HS;
  const float* kbase = qkv + (long)b * T * QKV_LD + N_HEAD * HS + kvh * HS;
  const float* vbase = kbase + N_KVHEAD * HS;   // +512

  const int col   = lane & 15;
  const int hh    = lane >> 4;
  const int half8 = hh * 8;

  // Q fragments: 16 rows x 128 dim = 4 x (16x32) bf16 A fragments
  v16bf qa[4];
#pragma unroll
  for (int d = 0; d < 4; ++d) qa[d] = load_frag(qbase + d * 32, QKV_LD, lane);

  v8f o[8];
#pragma unroll
  for (int d = 0; d < 8; ++d) o[d] = (v8f){};
  float mrow[8], lrow[8];
#pragma unroll
  for (int j = 0; j < 8; ++j) { mrow[j] = -__builtin_inff(); lrow[j] = 0.0f; }

  const float scale = 0.08838834764831845f;  // 1/sqrt(128)
  const int qend = qt * 16 + 15;
  const int nkb  = qend / 32 + 1;

  for (int kb = 0; kb < nkb; ++kb) {
    const int k0 = kb * 32;

    // prefetch next key block (lowers to global_prefetch_b8)
    if (kb + 1 < nkb)
      __builtin_prefetch(kbase + (long)(k0 + 32) * QKV_LD, 0, 1);

    // ---- scores: S = Q @ K^T for keys [k0, k0+32) ----
    v8f s0 = (v8f){}, s1 = (v8f){};
#pragma unroll
    for (int d = 0; d < 4; ++d) {
      v16bf bk0 = load_frag(kbase + (long)k0 * QKV_LD + d * 32, QKV_LD, lane);
      v16bf bk1 = load_frag(kbase + (long)(k0 + 16) * QKV_LD + d * 32, QKV_LD, lane);
      s0 = wmma_bf16(qa[d], bk0, s0);
      s1 = wmma_bf16(qa[d], bk1, s1);
    }

    // ---- scale + causal mask + online softmax ----
    float alpha[8];
#pragma unroll
    for (int j = 0; j < 8; ++j) {
      const int row = qt * 16 + half8 + j;
      float v0 = (k0 + col      <= row) ? s0[j] * scale : -__builtin_inff();
      float v1 = (k0 + 16 + col <= row) ? s1[j] * scale : -__builtin_inff();

      float mx = fmaxf(v0, v1);
#pragma unroll
      for (int off = 8; off >= 1; off >>= 1) mx = fmaxf(mx, __shfl_xor(mx, off, 32));
      const float mnew = fmaxf(mrow[j], mx);
      const float a  = __expf(mrow[j] - mnew);
      const float p0 = __expf(v0 - mnew);
      const float p1 = __expf(v1 - mnew);
      float rs = p0 + p1;
#pragma unroll
      for (int off = 8; off >= 1; off >>= 1) rs += __shfl_xor(rs, off, 32);
      lrow[j]  = lrow[j] * a + rs;
      mrow[j]  = mnew;
      alpha[j] = a;
      s0[j] = p0;
      s1[j] = p1;
    }

    // ---- P (C layout) -> LDS -> bf16 A fragment ----
#pragma unroll
    for (int j = 0; j < 8; ++j) {
      const int row = half8 + j;
      psh[row * 32 + col]      = (__bf16)s0[j];
      psh[row * 32 + 16 + col] = (__bf16)s1[j];
    }
    __syncthreads();
    v16bf pa;
    {
      const int r  = lane & 15;
      const int ka = hh * 8;
      const int kc = 16 + hh * 8;
#pragma unroll
      for (int i = 0; i < 8; ++i) pa[i]     = psh[r * 32 + ka + i];
#pragma unroll
      for (int i = 0; i < 8; ++i) pa[8 + i] = psh[r * 32 + kc + i];
    }
    __syncthreads();

    // ---- rescale O, then O += P @ V ----
#pragma unroll
    for (int dt = 0; dt < 8; ++dt)
#pragma unroll
      for (int j = 0; j < 8; ++j) o[dt][j] *= alpha[j];

#pragma unroll
    for (int dt = 0; dt < 8; ++dt) {
      // B fragment: V rows k0..k0+31 (K dim), cols dt*16..dt*16+15 (N dim).
      const float* vp = vbase + (long)k0 * QKV_LD + dt * 16 + col;
      v16bf vb;
#pragma unroll
      for (int i = 0; i < 8; ++i) vb[i]     = (__bf16)vp[(long)(i + hh * 8) * QKV_LD];
#pragma unroll
      for (int i = 0; i < 8; ++i) vb[8 + i] = (__bf16)vp[(long)(16 + i + hh * 8) * QKV_LD];
      o[dt] = wmma_bf16(pa, vb, o[dt]);
    }
  }

  // ---- epilogue: normalize by l and store [B,T,H,D] (== [B*T, 2048]) ----
  float* yb = y + ((long)b * T + (long)qt * 16) * MODEL + h * HS;
#pragma unroll
  for (int j = 0; j < 8; ++j) {
    const float rl = 1.0f / lrow[j];
    const int row  = half8 + j;
#pragma unroll
    for (int dt = 0; dt < 8; ++dt)
      yb[(long)row * MODEL + dt * 16 + col] = o[dt][j] * rl;
  }
}

// ---------------------------------------------------------------------------
extern "C" void kernel_launch(void* const* d_in, const int* in_sizes, int n_in,
                              void* d_out, int out_size, void* d_ws, size_t ws_size,
                              hipStream_t stream) {
  const float* x     = (const float*)d_in[0];  // [2,2048,2048]
  const float* w_qkv = (const float*)d_in[1];  // [3072,2048]
  const float* w_out = (const float*)d_in[2];  // [2048,2048]
  float* out = (float*)d_out;                  // [2,2048,2048]

  const int B = 2, T = 2048;
  const int MT = B * T;                        // 4096 rows

  // workspace: qkv [4096,3072] (50.3MB) + y_att [4096,2048] (33.5MB) = ~84MB
  float* qkv  = (float*)d_ws;
  float* yatt = qkv + (size_t)MT * QKV_LD;

  // 1) qkv = x @ w_qkv^T
  gemm_bf16_wmma<<<dim3(QKV_LD / 64, MT / 128), 128, 0, stream>>>(
      x, w_qkv, qkv, MT, QKV_LD, MODEL);

  // 2) RoPE in place on q,k sections
  {
    const long total = (long)MT * (N_HEAD + N_KVHEAD) * (HS / 2);
    const int  thr = 256;
    rope_kernel<<<(unsigned)((total + thr - 1) / thr), thr, 0, stream>>>(qkv, B, T);
  }

  // 3) flash attention -> yatt [B,T,16,128]
  attn_wmma_kernel<<<dim3(T / 16, N_HEAD, B), 32, 0, stream>>>(qkv, yatt, B, T);

  // 4) out = yatt @ w_out^T
  gemm_bf16_wmma<<<dim3(MODEL / 64, MT / 128), 128, 0, stream>>>(
      yatt, w_out, out, MT, MODEL, MODEL);
}
